// SIGEAttentionBlock3d_52905407152584
// MI455X (gfx1250) — matrix operand order repaired and checked by reading, not compiled, not run on hardware
//
#include <hip/hip_runtime.h>
#include <hip/hip_bf16.h>

// ---------------------------------------------------------------------------
// Types for CDNA5 WMMA (wave32): v_wmma_f32_16x16x32_bf16
// ---------------------------------------------------------------------------
typedef __bf16 bf16_t;
typedef __attribute__((ext_vector_type(16))) __bf16 v16bf;
typedef __attribute__((ext_vector_type(8)))  __bf16 v8bf;
typedef __attribute__((ext_vector_type(8)))  float  v8f;

#define DIM_C   512
#define NB      2
#define NT      16
#define NH      32
#define NW      32
#define BT      (NB * NT)       // 32
#define HW      (NH * NW)       // 1024
#define NTOK    (BT * HW)       // 32768
#define CHW     (NT * HW)       // channel stride inside x: 16384

// ---------------------------------------------------------------------------
// Async global->LDS copies (CDNA5 GLOBAL_LOAD_ASYNC_TO_LDS_B128, ASYNCcnt).
// Builtin signature (from hipcc diagnostic): param0 is
//   int __attribute__((vector_size(16))) __device__ *   (global int4*)
// Guarded: falls back to synchronous staging if the toolchain lacks them.
// ---------------------------------------------------------------------------
#if defined(__has_builtin)
#if __has_builtin(__builtin_amdgcn_global_load_async_to_lds_b128) && \
    __has_builtin(__builtin_amdgcn_s_wait_asynccnt)
#define CDNA5_ASYNC 1
#endif
#endif

#ifdef CDNA5_ASYNC
typedef int int4v __attribute__((vector_size(16)));
typedef __attribute__((address_space(1))) int4v g_int4v;
typedef __attribute__((address_space(3))) int4v l_int4v;
__device__ __forceinline__ void async_copy_b128(const void* g, void* l) {
    __builtin_amdgcn_global_load_async_to_lds_b128(
        (g_int4v*)g, (l_int4v*)l, /*offset=*/0, /*cpol=*/0);
}
#endif

// ---------------------------------------------------------------------------
// f32 -> bf16 elementwise conversion (weights)
// ---------------------------------------------------------------------------
__global__ void cvt_bf16_kernel(const float* __restrict__ src,
                                bf16_t* __restrict__ dst, int n) {
    int i = blockIdx.x * blockDim.x + threadIdx.x;
    if (i < n) dst[i] = (bf16_t)src[i];
}

// ---------------------------------------------------------------------------
// RMS norm over channel dim + layout change (b,c,t,h,w) -> (token, c) bf16.
// One wave32 per token; each lane owns 16 channels (stride-32 for coalescing).
// ---------------------------------------------------------------------------
__global__ __launch_bounds__(256) void rmsnorm_kernel(
        const float* __restrict__ x, const float* __restrict__ gamma,
        bf16_t* __restrict__ xn) {
    int tok  = (blockIdx.x * blockDim.x + threadIdx.x) >> 5;
    int lane = threadIdx.x & 31;
    if (tok >= NTOK) return;
    int bt  = tok >> 10;          // frame index
    int pix = tok & (HW - 1);
    int b = bt >> 4, t = bt & 15;
    size_t base = ((size_t)b * DIM_C * NT + t) * HW + pix;   // channel 0
    float vals[16];
    float ss = 0.f;
#pragma unroll
    for (int j = 0; j < 16; ++j) {
        int c = lane + 32 * j;
        float v = x[base + (size_t)c * CHW];
        vals[j] = v;
        ss += v * v;
    }
#pragma unroll
    for (int off = 16; off >= 1; off >>= 1) ss += __shfl_xor(ss, off, 32);
    float nrm   = sqrtf(ss);
    float scale = 22.62741699796952f / fmaxf(nrm, 1e-12f);   // sqrt(512)/nrm
#pragma unroll
    for (int j = 0; j < 16; ++j) {
        int c = lane + 32 * j;
        xn[(size_t)tok * DIM_C + c] = (bf16_t)(vals[j] * scale * gamma[c]);
    }
}

// ---------------------------------------------------------------------------
// Generic bf16 WMMA GEMM:  C[M,N] = A[M,K] * Bt[N,K]^T   (Bt row-major N x K)
// Block tile 128x128x32, 256 threads = 8 waves (4 x 2), wave tile 32x64.
// Double-buffered LDS; async global->LDS staging when available.
// MODE 0: qkv epilogue (split q/k/v, q pre-scaled, v stored transposed)
// MODE 1: plain bf16 store, batched via blockIdx.z
// MODE 2: proj epilogue (+bias, +residual from x-layout, f32 scatter out)
// ---------------------------------------------------------------------------
template<int MODE>
__global__ __launch_bounds__(256) void gemm_bf16_kernel(
        const bf16_t* __restrict__ A, const bf16_t* __restrict__ Bt,
        int N, int K,
        size_t strideA, size_t strideB, size_t strideC,
        bf16_t* __restrict__ out0, bf16_t* __restrict__ out1,
        bf16_t* __restrict__ out2,
        const float* __restrict__ bias,
        const float* __restrict__ resid, float* __restrict__ outf) {
    __shared__ bf16_t As[2][128][32];
    __shared__ bf16_t Bs[2][128][32];

    const int tid  = threadIdx.x;
    const int wid  = tid >> 5;
    const int lane = tid & 31;
    const int wrow = wid >> 1;          // 0..3  -> 32-row band
    const int wcol = wid & 1;           // 0..1  -> 64-col band
    const int fr   = lane & 15;         // row/col within 16x16 tile
    const int hi   = lane >> 4;         // lane half

    const int m0 = blockIdx.y * 128;
    const int n0 = blockIdx.x * 128;
    const int z  = blockIdx.z;
    const bf16_t* Ap = A  + (size_t)z * strideA;
    const bf16_t* Bp = Bt + (size_t)z * strideB;

    v8f acc[2][4];
#pragma unroll
    for (int i = 0; i < 2; ++i)
#pragma unroll
        for (int j = 0; j < 4; ++j)
#pragma unroll
            for (int r = 0; r < 8; ++r) acc[i][j][r] = 0.f;

    const int lrow = tid >> 1;          // 0..127: staged tile row
    const int lk   = (tid & 1) * 16;    // 0 or 16: K half

    const bf16_t* arow = Ap + (size_t)(m0 + lrow) * K + lk;
    const bf16_t* brow = Bp + (size_t)(n0 + lrow) * K + lk;
    const int nsteps = K >> 5;

#ifdef CDNA5_ASYNC
    // prologue: stage K-slab 0 into buffer 0 (4 async b128 per lane)
    async_copy_b128(arow,     &As[0][lrow][lk]);
    async_copy_b128(arow + 8, &As[0][lrow][lk + 8]);
    async_copy_b128(brow,     &Bs[0][lrow][lk]);
    async_copy_b128(brow + 8, &Bs[0][lrow][lk + 8]);
#endif

    for (int i = 0; i < nsteps; ++i) {
        const int cur = i & 1;
#ifdef CDNA5_ASYNC
        if (i + 1 < nsteps) {
            // overlap: issue next slab into the other buffer, then wait only
            // for the current slab's 4 copies (ASYNCcnt completes in order).
            const bf16_t* an = arow + (size_t)(i + 1) * 32;
            const bf16_t* bn = brow + (size_t)(i + 1) * 32;
            async_copy_b128(an,     &As[cur ^ 1][lrow][lk]);
            async_copy_b128(an + 8, &As[cur ^ 1][lrow][lk + 8]);
            async_copy_b128(bn,     &Bs[cur ^ 1][lrow][lk]);
            async_copy_b128(bn + 8, &Bs[cur ^ 1][lrow][lk + 8]);
            __builtin_amdgcn_s_wait_asynccnt(4);
        } else {
            __builtin_amdgcn_s_wait_asynccnt(0);
        }
        __syncthreads();
#else
        {
            const int k0 = i * 32;
            const uint4* ag = (const uint4*)(arow + k0);
            uint4 a0 = ag[0], a1 = ag[1];
            *((uint4*)&As[cur][lrow][lk])     = a0;
            *((uint4*)&As[cur][lrow][lk + 8]) = a1;
            const uint4* bg = (const uint4*)(brow + k0);
            uint4 b0 = bg[0], b1 = bg[1];
            *((uint4*)&Bs[cur][lrow][lk])     = b0;
            *((uint4*)&Bs[cur][lrow][lk + 8]) = b1;
            if (k0 + 32 < K)   // hint next slab toward L2 (global_prefetch_b8)
                __builtin_prefetch(arow + k0 + 32, 0, 3);
        }
        __syncthreads();
#endif

        // ---- load fragments per ISA 16-bit A(16x32)/B(32x16) layouts ----
        v16bf afrag[2];
#pragma unroll
        for (int mt = 0; mt < 2; ++mt) {
            int r = wrow * 32 + mt * 16 + fr;
            union { v16bf v; v8bf h[2]; } u;
            u.h[0] = *(const v8bf*)&As[cur][r][hi * 8];        // K 0-7 / 8-15
            u.h[1] = *(const v8bf*)&As[cur][r][16 + hi * 8];   // K 16-23 / 24-31
            afrag[mt] = u.v;
        }
        v16bf bfrag[4];
#pragma unroll
        for (int nt = 0; nt < 4; ++nt) {
            int n = wcol * 64 + nt * 16 + fr;
            union { v16bf v; v8bf h[2]; } u;
            u.h[0] = *(const v8bf*)&Bs[cur][n][hi * 8];        // K 0-7 / 8-15
            u.h[1] = *(const v8bf*)&Bs[cur][n][16 + hi * 8];   // K 16-23 / 24-31
            bfrag[nt] = u.v;
        }

        // ---- 8 WMMAs per K-step ----
#pragma unroll
        for (int mt = 0; mt < 2; ++mt)
#pragma unroll
            for (int nt = 0; nt < 4; ++nt)
                acc[mt][nt] = __builtin_amdgcn_wmma_f32_16x16x32_bf16(
                    false, afrag[mt], false, bfrag[nt],
                    (short)0, acc[mt][nt], false, false);
        __syncthreads();
    }

    // ---- epilogue: C tile layout VGPR r -> M = r + 8*hi, N = fr ----
#pragma unroll
    for (int mt = 0; mt < 2; ++mt)
#pragma unroll
        for (int nt = 0; nt < 4; ++nt)
#pragma unroll
            for (int r = 0; r < 8; ++r) {
                int m = m0 + wrow * 32 + mt * 16 + (hi << 3) + r;
                int n = n0 + wcol * 64 + nt * 16 + fr;
                float v = acc[mt][nt][r];
                if (MODE == 0) {                 // qkv split
                    v += bias[n];
                    if (n < DIM_C) {             // q, fold attention scale
                        out0[(size_t)m * DIM_C + n] =
                            (bf16_t)(v * 0.04419417382415922f);  // 512^-0.5
                    } else if (n < 2 * DIM_C) {  // k
                        out1[(size_t)m * DIM_C + (n - DIM_C)] = (bf16_t)v;
                    } else {                     // v, stored transposed [ch][tok]
                        int b_  = m >> 10;
                        int row = m & (HW - 1);
                        int ch  = n - 2 * DIM_C;
                        out2[((size_t)b_ * DIM_C + ch) * HW + row] = (bf16_t)v;
                    }
                } else if (MODE == 1) {          // plain batched bf16 store
                    out0[(size_t)z * strideC + (size_t)m * N + n] = (bf16_t)v;
                } else {                         // proj + residual + scatter
                    v += bias[n];
                    int bt  = m >> 10;
                    int pix = m & (HW - 1);
                    int b_ = bt >> 4, t_ = bt & 15;
                    size_t idx = (((size_t)b_ * DIM_C + n) * NT + t_) * HW + pix;
                    outf[idx] = v + resid[idx];
                }
            }
}

// ---------------------------------------------------------------------------
// Row softmax over 1024-wide rows, bf16 in-place (f32 math).
// ---------------------------------------------------------------------------
__global__ __launch_bounds__(256) void softmax_kernel(bf16_t* __restrict__ s) {
    __shared__ float red[256];
    bf16_t* p  = s + (size_t)blockIdx.x * HW;
    int tid = threadIdx.x;
    float v[4];
    float mx = -3.0e38f;
#pragma unroll
    for (int j = 0; j < 4; ++j) {
        v[j] = (float)p[tid + 256 * j];
        mx = fmaxf(mx, v[j]);
    }
    red[tid] = mx; __syncthreads();
    for (int o = 128; o >= 1; o >>= 1) {
        if (tid < o) red[tid] = fmaxf(red[tid], red[tid + o]);
        __syncthreads();
    }
    mx = red[0]; __syncthreads();
    float sm = 0.f;
#pragma unroll
    for (int j = 0; j < 4; ++j) { v[j] = __expf(v[j] - mx); sm += v[j]; }
    red[tid] = sm; __syncthreads();
    for (int o = 128; o >= 1; o >>= 1) {
        if (tid < o) red[tid] += red[tid + o];
        __syncthreads();
    }
    float inv = 1.f / red[0];
#pragma unroll
    for (int j = 0; j < 4; ++j) p[tid + 256 * j] = (bf16_t)(v[j] * inv);
}

// ---------------------------------------------------------------------------
// Launch: inputs = x, gamma, qkv_w, qkv_b, proj_w, proj_b  (all f32)
// ---------------------------------------------------------------------------
extern "C" void kernel_launch(void* const* d_in, const int* in_sizes, int n_in,
                              void* d_out, int out_size, void* d_ws, size_t ws_size,
                              hipStream_t stream) {
    const float* x      = (const float*)d_in[0];
    const float* gamma  = (const float*)d_in[1];
    const float* qkv_w  = (const float*)d_in[2];
    const float* qkv_b  = (const float*)d_in[3];
    const float* proj_w = (const float*)d_in[4];
    const float* proj_b = (const float*)d_in[5];
    float* out = (float*)d_out;

    // workspace carve-up (all section sizes multiple of 256B)
    char* w = (char*)d_ws;
    bf16_t* wq   = (bf16_t*)w; w += (size_t)3 * DIM_C * DIM_C * 2;      // 1.5 MB
    bf16_t* wp   = (bf16_t*)w; w += (size_t)DIM_C * DIM_C * 2;          // 0.5 MB
    bf16_t* xn   = (bf16_t*)w; w += (size_t)NTOK * DIM_C * 2;           // 32 MB
    bf16_t* qbuf = (bf16_t*)w; w += (size_t)NTOK * DIM_C * 2;           // 32 MB
    bf16_t* kbuf = (bf16_t*)w; w += (size_t)NTOK * DIM_C * 2;           // 32 MB
    bf16_t* vt   = (bf16_t*)w; w += (size_t)NTOK * DIM_C * 2;           // 32 MB
    bf16_t* sbuf = (bf16_t*)w; w += (size_t)BT * HW * HW * 2;           // 64 MB
    bf16_t* obuf = (bf16_t*)w; w += (size_t)NTOK * DIM_C * 2;           // 32 MB

    dim3 blk(256);

    // 0) weights -> bf16
    cvt_bf16_kernel<<<(3 * DIM_C * DIM_C + 255) / 256, blk, 0, stream>>>(qkv_w, wq, 3 * DIM_C * DIM_C);
    cvt_bf16_kernel<<<(DIM_C * DIM_C + 255) / 256, blk, 0, stream>>>(proj_w, wp, DIM_C * DIM_C);

    // 1) RMS norm + transpose -> xn (token x 512) bf16
    rmsnorm_kernel<<<NTOK / 8, blk, 0, stream>>>(x, gamma, xn);

    // 2) QKV GEMM: (32768 x 1536) = xn (32768 x 512) * qkv_w^T
    gemm_bf16_kernel<0><<<dim3(1536 / 128, NTOK / 128, 1), blk, 0, stream>>>(
        xn, wq, 1536, DIM_C, 0, 0, 0, qbuf, kbuf, vt, qkv_b, nullptr, nullptr);

    // 3) S = Q * K^T per frame: (1024 x 1024), K=512, 32 batches
    gemm_bf16_kernel<1><<<dim3(HW / 128, HW / 128, BT), blk, 0, stream>>>(
        qbuf, kbuf, HW, DIM_C,
        (size_t)HW * DIM_C, (size_t)HW * DIM_C, (size_t)HW * HW,
        sbuf, nullptr, nullptr, nullptr, nullptr, nullptr);

    // 4) softmax rows (32768 rows of 1024)
    softmax_kernel<<<BT * HW, blk, 0, stream>>>(sbuf);

    // 5) O = P * V per frame: (1024 x 512), K=1024, Bt = vt[ch][tok]
    gemm_bf16_kernel<1><<<dim3(DIM_C / 128, HW / 128, BT), blk, 0, stream>>>(
        sbuf, vt, DIM_C, HW,
        (size_t)HW * HW, (size_t)DIM_C * HW, (size_t)HW * DIM_C,
        obuf, nullptr, nullptr, nullptr, nullptr, nullptr);

    // 6) proj + bias + residual + scatter to (b,c,t,h,w)
    gemm_bf16_kernel<2><<<dim3(DIM_C / 128, NTOK / 128, 1), blk, 0, stream>>>(
        obuf, wp, DIM_C, DIM_C, 0, 0, 0,
        nullptr, nullptr, nullptr, proj_b, x, out);
}
